// SoftMoE_8383776162603
// MI455X (gfx1250) — matrix operand order repaired and checked by reading, not compiled, not run on hardware
//
#include <hip/hip_runtime.h>
#include <hip/hip_bf16.h>
#include <math.h>

// SoftMoE forward for MI455X (gfx1250, wave32).
// All 4 contractions run through v_wmma_f32_16x16x32_bf16 (fp32 memory, bf16
// RNE operands, fp32 accumulate). ~0.7 GB compulsory DRAM traffic vs 120 GFLOP
// => memory-bound; staging is vectorized (global_load_b128 -> ds_store_b128),
// softmaxes are single-read (register-cached). Tile re-reads across block
// columns hit the 192 MB L2 (panels are 8-32 MB).
//
// Workspace (floats), 192 MB total:
//   ws0 [16,777,216] : logits (b,m,n*p) -> reused later: slots (32MB) + y (32MB)
//   ws1 [16,777,216] : dispatch
//   ws2 [16,777,216] : combine

typedef __attribute__((ext_vector_type(16))) __bf16       bf16x16;
typedef __attribute__((ext_vector_type(8)))  float        f32x8;
typedef __attribute__((ext_vector_type(4)))  unsigned int u32x4;

static constexpr int BM = 128;       // block tile M
static constexpr int BN = 64;        // block tile N
static constexpr int BK = 32;        // K step (== WMMA K)
static constexpr int NTHREADS = 256; // 8 wave32 waves: 4 (M) x 2 (N)

__device__ __forceinline__ unsigned short f32_to_bf16_rne(float f) {
    unsigned u = __float_as_uint(f);
    u += 0x7FFFu + ((u >> 16) & 1u);
    return (unsigned short)(u >> 16);
}

#if defined(__has_builtin)
#if __has_builtin(__builtin_amdgcn_cvt_pk_bf16_f32)
#define HAVE_CVT_PK_BF16 1
#endif
#endif

__device__ __forceinline__ unsigned pack_bf16x2(float lo, float hi) {
#if HAVE_CVT_PK_BF16
    typedef __attribute__((ext_vector_type(2))) __bf16 bf16x2;
    bf16x2 p = __builtin_amdgcn_cvt_pk_bf16_f32(lo, hi);
    return __builtin_bit_cast(unsigned, p);
#else
    return (unsigned)f32_to_bf16_rne(lo) | ((unsigned)f32_to_bf16_rne(hi) << 16);
#endif
}

union Frag { bf16x16 v; u32x4 q[2]; };

// C = A x B (+ bias[col]); B row-major (K_ x N_), C row-major (M_ x N_).
// AMODE 0: A row-major, element (i,k) = A[i*lda + k]
// AMODE 1: A col-major, element (i,k) = A[k*lda + i]   (free transpose)
// MGUARD: bounds-check rows (only needed when M_ % BM != 0).
// Requires: K_ % BK == 0, N_ % BN == 0, 16B alignment of all tiles.
template<int AMODE, bool MGUARD>
__global__ __launch_bounds__(NTHREADS)
void gemm_bf16_wmma(const float* __restrict__ A, const float* __restrict__ B,
                    float* __restrict__ C, const float* __restrict__ bias,
                    int M_, int N_, int K_,
                    long long lda, long long aBatch,
                    long long ldb, long long bBatch, int bMod,
                    long long ldc, long long cBatch,
                    long long biasBatch, int biasMod)
{
    __shared__ unsigned short ldsA[BM * BK];    // 8 KB, [m][k]
    __shared__ unsigned short ldsBt[BN * BK];   // 4 KB, B transposed: [n][k]

    const int tid  = threadIdx.x;
    const int lane = tid & 31;
    const int half = lane >> 4;
    const int wave = tid >> 5;
    const int wm   = wave >> 1;    // 0..3
    const int wn   = wave & 1;     // 0..1

    const int batch = blockIdx.z;
    const float* Ab = A + (long long)batch * aBatch;
    const float* Bb = B + (long long)(batch % bMod) * bBatch;
    float*       Cb = C + (long long)batch * cBatch;
    const float* biasb = bias ? (bias + (long long)(batch % biasMod) * biasBatch) : nullptr;

    const int row0 = blockIdx.y * BM;
    const int col0 = blockIdx.x * BN;

    f32x8 acc[2][2];
    #pragma unroll
    for (int i = 0; i < 2; ++i)
        #pragma unroll
        for (int j = 0; j < 2; ++j)
            #pragma unroll
            for (int e = 0; e < 8; ++e) acc[i][j][e] = 0.0f;

    for (int k0 = 0; k0 < K_; k0 += BK) {
        // ---------- stage A tile (BM x BK bf16) ----------
        if (AMODE == 0) {
            // contiguous along K: thread -> row r, 16 floats = 4x b128 loads
            const int r  = tid >> 1;            // 0..127
            const int kb = (tid & 1) * 16;
            unsigned short* dst = &ldsA[r * BK + kb];
            u32x4 p0 = {0u, 0u, 0u, 0u}, p1 = {0u, 0u, 0u, 0u};
            if (!MGUARD || (row0 + r) < M_) {
                const float4* s4 = reinterpret_cast<const float4*>(
                    Ab + (long long)(row0 + r) * lda + (k0 + kb));
                const float4 f0 = s4[0], f1 = s4[1], f2 = s4[2], f3 = s4[3];
                p0 = u32x4{ pack_bf16x2(f0.x, f0.y), pack_bf16x2(f0.z, f0.w),
                            pack_bf16x2(f1.x, f1.y), pack_bf16x2(f1.z, f1.w) };
                p1 = u32x4{ pack_bf16x2(f2.x, f2.y), pack_bf16x2(f2.z, f2.w),
                            pack_bf16x2(f3.x, f3.y), pack_bf16x2(f3.z, f3.w) };
                if (k0 + BK < K_)  // gfx1250 global_prefetch_b8, next K tile
                    __builtin_prefetch(reinterpret_cast<const float*>(s4) + BK, 0, 0);
            }
            *reinterpret_cast<u32x4*>(dst)     = p0;   // ds_store_b128
            *reinterpret_cast<u32x4*>(dst + 8) = p1;   // ds_store_b128
        } else {
            // contiguous along I: thread -> k-row, 16 contiguous i
            const int k  = tid >> 3;            // 0..31
            const int ib = (tid & 7) * 16;
            const float4* s4 = reinterpret_cast<const float4*>(
                Ab + (long long)(k0 + k) * lda + (row0 + ib));
            const float4 f0 = s4[0], f1 = s4[1], f2 = s4[2], f3 = s4[3];
            const float fv[16] = { f0.x,f0.y,f0.z,f0.w, f1.x,f1.y,f1.z,f1.w,
                                   f2.x,f2.y,f2.z,f2.w, f3.x,f3.y,f3.z,f3.w };
            #pragma unroll
            for (int ii = 0; ii < 16; ++ii) {
                const float v = (!MGUARD || (row0 + ib + ii) < M_) ? fv[ii] : 0.0f;
                ldsA[(ib + ii) * BK + k] = f32_to_bf16_rne(v);
            }
            if (k0 + BK < K_)
                __builtin_prefetch(reinterpret_cast<const float*>(s4) + BK * lda, 0, 0);
        }
        // ---------- stage B tile transposed (BN x BK bf16) ----------
        {
            // B is (K x N) row-major; contiguous along N: thread -> k-row, 8 n
            const int k  = tid >> 3;            // 0..31
            const int n0 = (tid & 7) * 8;
            const float4* s4 = reinterpret_cast<const float4*>(
                Bb + (long long)(k0 + k) * ldb + (col0 + n0));
            const float4 f0 = s4[0], f1 = s4[1];
            const float fv[8] = { f0.x,f0.y,f0.z,f0.w, f1.x,f1.y,f1.z,f1.w };
            #pragma unroll
            for (int j = 0; j < 8; ++j)
                ldsBt[(n0 + j) * BK + k] = f32_to_bf16_rne(fv[j]);
            if (k0 + BK < K_)
                __builtin_prefetch(reinterpret_cast<const float*>(s4) + BK * ldb, 0, 0);
        }
        __syncthreads();

        // ---------- fragments per ISA VGPR layouts ----------
        Frag fa[2], fb[2];
        #pragma unroll
        for (int i = 0; i < 2; ++i) {
            // A 16-bit 16x32: lane holds row m=lane%16; K = {8h..8h+7, 16+8h..+7}
            const int m = wm * 32 + i * 16 + (lane & 15);
            const u32x4* rp = reinterpret_cast<const u32x4*>(&ldsA[m * BK]);
            fa[i].q[0] = rp[half];
            fa[i].q[1] = rp[2 + half];
        }
        #pragma unroll
        for (int j = 0; j < 2; ++j) {
            // B 16-bit 32x16: lane holds col n=lane%16; K = {8h..8h+7, 16+8h..+7}
            const int n = wn * 32 + j * 16 + (lane & 15);
            const u32x4* rp = reinterpret_cast<const u32x4*>(&ldsBt[n * BK]);
            fb[j].q[0] = rp[half];
            fb[j].q[1] = rp[2 + half];
        }
        #pragma unroll
        for (int i = 0; i < 2; ++i)
            #pragma unroll
            for (int j = 0; j < 2; ++j)
                acc[i][j] = __builtin_amdgcn_wmma_f32_16x16x32_bf16(
                    false, fa[i].v, false, fb[j].v, (short)0, acc[i][j], false, false);
        __syncthreads();
    }

    // ---------- epilogue: VGPR r -> M = r + 8*half, N = lane%16 ----------
    #pragma unroll
    for (int i = 0; i < 2; ++i) {
        #pragma unroll
        for (int j = 0; j < 2; ++j) {
            const int gc = col0 + wn * 32 + j * 16 + (lane & 15);
            const float bv = biasb ? biasb[gc] : 0.0f;
            #pragma unroll
            for (int r = 0; r < 8; ++r) {
                const int gr = row0 + wm * 32 + i * 16 + r + 8 * half;
                if (!MGUARD || gr < M_)
                    Cb[(long long)gr * ldc + gc] = acc[i][j][r] + bv;
            }
        }
    }
}

// softmax over the batch axis (size 4): one thread per (m,n,p); single read.
__global__ void softmax_over_batch4(const float* __restrict__ logits, float* __restrict__ out,
                                    int total, long long bstride)
{
    const int idx = blockIdx.x * blockDim.x + threadIdx.x;
    if (idx >= total) return;
    float v[4];
    #pragma unroll
    for (int b = 0; b < 4; ++b) v[b] = logits[(long long)b * bstride + idx];
    float mx = fmaxf(fmaxf(v[0], v[1]), fmaxf(v[2], v[3]));
    float e[4], sum = 0.0f;
    #pragma unroll
    for (int b = 0; b < 4; ++b) { e[b] = __expf(v[b] - mx); sum += e[b]; }
    const float inv = 1.0f / sum;
    #pragma unroll
    for (int b = 0; b < 4; ++b) out[(long long)b * bstride + idx] = e[b] * inv;
}

// row softmax over ncols (<= 256*VPT): one 256-thread block per (b,m) row;
// values cached in registers so the row is read exactly once.
template<int VPT>
__global__ __launch_bounds__(256)
void softmax_rows(const float* __restrict__ in, float* __restrict__ out, int ncols)
{
    __shared__ float red[256];
    const long long base = (long long)blockIdx.x * ncols;
    const int tid = threadIdx.x;

    float v[VPT];
    float mx = -3.0e38f;
    #pragma unroll
    for (int u = 0; u < VPT; ++u) {
        const int c = tid + u * 256;
        v[u] = (c < ncols) ? in[base + c] : -3.0e38f;
        mx = fmaxf(mx, v[u]);
    }
    red[tid] = mx; __syncthreads();
    for (int s = 128; s > 0; s >>= 1) {
        if (tid < s) red[tid] = fmaxf(red[tid], red[tid + s]);
        __syncthreads();
    }
    mx = red[0]; __syncthreads();

    float sum = 0.0f;
    #pragma unroll
    for (int u = 0; u < VPT; ++u) {
        v[u] = __expf(v[u] - mx);
        sum += v[u];
    }
    red[tid] = sum; __syncthreads();
    for (int s = 128; s > 0; s >>= 1) {
        if (tid < s) red[tid] += red[tid + s];
        __syncthreads();
    }
    const float inv = 1.0f / red[0];
    #pragma unroll
    for (int u = 0; u < VPT; ++u) {
        const int c = tid + u * 256;
        if (c < ncols) out[base + c] = v[u] * inv;
    }
}

extern "C" void kernel_launch(void* const* d_in, const int* in_sizes, int n_in,
                              void* d_out, int out_size, void* d_ws, size_t ws_size,
                              hipStream_t stream)
{
    const float* x    = (const float*)d_in[0];   // (4, 2048, 1024)
    const float* phi  = (const float*)d_in[1];   // (1024, 64, 32) == (1024, 2048)
    const float* W    = (const float*)d_in[2];   // (64, 1024, 1024)
    const float* bias = (const float*)d_in[3];   // (64, 1024)
    float* out = (float*)d_out;                  // (4, 2048, 1024)

    const int Bz = 4, M = 2048, D = 1024, N = 64, P = 32;
    const int NP = N * P;                          // 2048
    const long long lgSz = (long long)Bz * M * NP; // 16,777,216 floats (64 MB)

    float* logits   = (float*)d_ws;        // ws0
    float* dispatch = logits + lgSz;       // ws1
    float* combine  = dispatch + lgSz;     // ws2
    float* slots    = logits;                          // reuse ws0[0 : 32MB]
    float* y        = logits + (long long)Bz * NP * D; // reuse ws0[32MB : 64MB]

    const dim3 blk(NTHREADS);

    // 1) logits(8192 x 2048) = x(8192 x 1024) @ phi(1024 x 2048)
    gemm_bf16_wmma<0, false><<<dim3(NP / BN, (Bz * M) / BM, 1), blk, 0, stream>>>(
        x, phi, logits, nullptr,
        Bz * M, NP, D,
        /*lda*/ D, /*aBatch*/ 0,
        /*ldb*/ NP, /*bBatch*/ 0, /*bMod*/ 1,
        /*ldc*/ NP, /*cBatch*/ 0,
        0, 1);

    // 2) dispatch = softmax over batch axis (4 values per (m,n,p))
    const int total = M * NP;
    softmax_over_batch4<<<(total + 255) / 256, blk, 0, stream>>>(
        logits, dispatch, total, (long long)M * NP);

    // 3) combine = softmax over the 2048-slot axis, per (b,m)
    softmax_rows<8><<<Bz * M, blk, 0, stream>>>(logits, combine, NP);

    // 4) slots[b](2048 x 1024) = dispatch[b]^T (NP x M) @ x[b] (M x D)
    //    A col-major: element (i,k) = dispatch[b][k*NP + i]
    gemm_bf16_wmma<1, false><<<dim3(D / BN, NP / BM, Bz), blk, 0, stream>>>(
        dispatch, x, slots, nullptr,
        NP, D, M,
        /*lda*/ NP, /*aBatch*/ (long long)M * NP,
        /*ldb*/ D, /*bBatch*/ (long long)M * D, /*bMod*/ Bz,
        /*ldc*/ D, /*cBatch*/ (long long)NP * D,
        0, 1);

    // 5) y[b,n](32 x 1024) = slots[b,n](32 x 1024) @ W[n](1024 x 1024) + b[n]
    gemm_bf16_wmma<0, true><<<dim3(D / BN, 1, Bz * N), blk, 0, stream>>>(
        slots, W, y, bias,
        P, D, D,
        /*lda*/ D, /*aBatch*/ (long long)P * D,
        /*ldb*/ D, /*bBatch*/ (long long)D * D, /*bMod*/ N,
        /*ldc*/ D, /*cBatch*/ (long long)P * D,
        /*biasBatch*/ (long long)D, /*biasMod*/ N);

    // 6) out[b](2048 x 1024) = combine[b](M x NP) @ y[b](NP x D)
    gemm_bf16_wmma<0, false><<<dim3(D / BN, M / BM, Bz), blk, 0, stream>>>(
        combine, y, out, nullptr,
        M, D, NP,
        /*lda*/ NP, /*aBatch*/ (long long)M * NP,
        /*ldb*/ D, /*bBatch*/ (long long)NP * D, /*bMod*/ Bz,
        /*ldc*/ D, /*cBatch*/ (long long)M * D,
        0, 1);
}
